// Attension_Layer_72370198938109
// MI455X (gfx1250) — matrix operand order repaired and checked by reading, not compiled, not run on hardware
//
#include <hip/hip_runtime.h>
#include <hip/hip_bf16.h>

// MI455X / gfx1250, wave32. All matmuls via v_wmma_f32_16x16x32_bf16.
// Fused batch-softmax attention: energy never hits HBM (saves ~270MB traffic).

typedef __bf16 bf16_t;
typedef bf16_t v16bf __attribute__((ext_vector_type(16)));
typedef bf16_t v8bf  __attribute__((ext_vector_type(8)));
typedef float  v8f   __attribute__((ext_vector_type(8)));

#define BS   8
#define CHN  256
#define CR32 32
#define C2   128
#define LL   4096
#define L4   1024
#define NCAT 192   // 32 theta + 32 phi + 128 g output channels

// ---- WMMA operand loaders (per CDNA5 ISA 7.12.2 layouts) -------------------
// A 16x32 bf16: lane<16 -> row=lane, K{0..7,16..23}; lane>=16 -> row=lane-16, K{8..15,24..31}
__device__ __forceinline__ v16bf load_a16x32(const bf16_t* base, int ld, int lane) {
  const bf16_t* p = base + (lane & 15) * ld + ((lane < 16) ? 0 : 8);
  v8bf lo = *(const v8bf*)(p);
  v8bf hi = *(const v8bf*)(p + 16);
  return __builtin_shufflevector(lo, hi, 0,1,2,3,4,5,6,7,8,9,10,11,12,13,14,15);
}
// B 32x16 bf16 stored as B[k][n] = base[n*ldn + k]: lane<16 -> col=lane, K{0..7,16..23}; lane>=16 -> col=lane-16, K{8..15,24..31}
__device__ __forceinline__ v16bf load_b32x16(const bf16_t* base, int ldn, int lane) {
  const bf16_t* p = base + (lane & 15) * ldn + ((lane < 16) ? 0 : 8);
  v8bf lo = *(const v8bf*)(p);
  v8bf hi = *(const v8bf*)(p + 16);
  return __builtin_shufflevector(lo, hi, 0,1,2,3,4,5,6,7,8,9,10,11,12,13,14,15);
}
__device__ __forceinline__ v8f wmma_bf16(v16bf a, v16bf b, v8f c) {
  return __builtin_amdgcn_wmma_f32_16x16x32_bf16(false, a, false, b, (short)0, c, false, false);
}

// ---- k0: x [b][c][l] f32 -> xT [b][l][c] bf16 ------------------------------
__global__ void __launch_bounds__(256) k_transpose_x(const float* __restrict__ x,
                                                     bf16_t* __restrict__ xT) {
  __shared__ float tile[32][33];
  int blk = blockIdx.x;            // 8192 = 8 b * 8 ctiles * 128 ltiles
  int b   = blk >> 10;
  int r   = blk & 1023;
  int ct0 = (r >> 7) << 5;
  int lt0 = (r & 127) << 5;
  int t = threadIdx.x, tx = t & 31, ty = t >> 5;
  const float* xb = x + (size_t)b * CHN * LL;
  #pragma unroll
  for (int k = 0; k < 4; ++k) {
    int c = ty + 8 * k;
    tile[c][tx] = xb[(size_t)(ct0 + c) * LL + lt0 + tx];
  }
  __syncthreads();
  bf16_t* xo = xT + (size_t)b * LL * CHN;
  #pragma unroll
  for (int k = 0; k < 4; ++k) {
    int lr = ty + 8 * k;
    xo[(size_t)(lt0 + lr) * CHN + ct0 + tx] = (bf16_t)tile[tx][lr];
  }
}

// ---- k0w: pack weights to bf16, concat projection weights ------------------
__global__ void k_prep_weights(const float* __restrict__ wt, const float* __restrict__ wp,
                               const float* __restrict__ wg, const float* __restrict__ bt,
                               const float* __restrict__ bp, const float* __restrict__ bg,
                               const float* __restrict__ wl,
                               bf16_t* __restrict__ wcat, bf16_t* __restrict__ wlastb,
                               float* __restrict__ bcat) {
  int t = threadIdx.x;
  for (int i = t; i < 32 * 256;  i += 256) wcat[i]            = (bf16_t)wt[i];
  for (int i = t; i < 32 * 256;  i += 256) wcat[32 * 256 + i] = (bf16_t)wp[i];
  for (int i = t; i < 128 * 256; i += 256) wcat[64 * 256 + i] = (bf16_t)wg[i];
  for (int i = t; i < 256 * 128; i += 256) wlastb[i]          = (bf16_t)wl[i];
  if (t < 32)  { bcat[t] = bt[t]; bcat[32 + t] = bp[t]; }
  if (t < 128) bcat[64 + t] = bg[t];
}

// ---- k1: fused theta/phi/g 1x1 convs (WMMA) + maxpool epilogue -------------
// Block covers one image row-pair (128 pixels) for one batch: 192x128 conv tile.
__global__ void __launch_bounds__(256) k_proj(const bf16_t* __restrict__ xT,
                                              const bf16_t* __restrict__ wcat,
                                              const float* __restrict__ bcat,
                                              bf16_t* __restrict__ thetaT,
                                              bf16_t* __restrict__ phiT,
                                              bf16_t* __restrict__ gT) {
  __shared__ float conv[NCAT][129];   // +1 pad vs 64-bank LDS
  int b  = blockIdx.x >> 5;           // 8
  int rp = blockIdx.x & 31;           // 32 row-pairs
  int l0 = rp * 128;
  int t = threadIdx.x, w = t >> 5, lane = t & 31;
  const bf16_t* xb = xT + ((size_t)b * LL + l0) * CHN;
  for (int i = 0; i < 12; ++i) {      // 96 (M,N) tile pairs over 8 waves
    int q = w * 12 + i;
    int mt = q >> 3, nt = q & 7;
    v8f acc = {};
    #pragma unroll
    for (int kc = 0; kc < 8; ++kc) {  // K = 256
      v16bf A = load_a16x32(wcat + (size_t)mt * 16 * CHN + kc * 32, CHN, lane);
      v16bf B = load_b32x16(xb + (size_t)nt * 16 * CHN + kc * 32, CHN, lane);
      acc = wmma_bf16(A, B, acc);
    }
    int col = nt * 16 + (lane & 15);
    int r0  = mt * 16 + ((lane < 16) ? 0 : 8);
    #pragma unroll
    for (int v = 0; v < 8; ++v)
      conv[r0 + v][col] = acc[v] + bcat[r0 + v];
  }
  __syncthreads();
  // theta: channels 0..31 -> thetaT[b][l][d]
  for (int i = t; i < 128 * 32; i += 256) {
    int d = i & 31, pr = i >> 5;
    thetaT[((size_t)b * LL + l0 + pr) * 32 + d] = (bf16_t)conv[d][pr];
  }
  int m0 = rp * 32;
  // phi: channels 32..63, maxpool 2x2 -> phiT[b][m][d]
  for (int i = t; i < 32 * 32; i += 256) {
    int d = i & 31, pw = i >> 5, p0 = 2 * pw;
    float mx = fmaxf(fmaxf(conv[32 + d][p0], conv[32 + d][p0 + 1]),
                     fmaxf(conv[32 + d][64 + p0], conv[32 + d][64 + p0 + 1]));
    phiT[((size_t)b * L4 + m0 + pw) * 32 + d] = (bf16_t)mx;
  }
  // g: channels 64..191, maxpool, stored so the raw [L4][C2] view reads as gT[b][c'][m']
  for (int i = t; i < 32 * 128; i += 256) {
    int c = i & 127, pw = i >> 7, p0 = 2 * pw;
    float mx = fmaxf(fmaxf(conv[64 + c][p0], conv[64 + c][p0 + 1]),
                     fmaxf(conv[64 + c][64 + p0], conv[64 + c][64 + p0 + 1]));
    int f = c * L4 + (m0 + pw);       // flat index of conv-layout [c][m]
    gT[(size_t)b * (C2 * L4) + (size_t)(f & 127) * L4 + (f >> 7)] = (bf16_t)mx;
  }
}

// ---- k2: fused energy -> softmax(axis=batch) -> attn store + attn@g --------
// One block = one 16-row l-tile; wave w == batch w (softmax couples batches).
__global__ void __launch_bounds__(256) k_attn(const bf16_t* __restrict__ thetaT,
                                              const bf16_t* __restrict__ phiT,
                                              const bf16_t* __restrict__ gT,
                                              float* __restrict__ attn_out,   // [8][4096][1024]
                                              bf16_t* __restrict__ avT) {     // [8][4096][128] view-transposed
  __shared__ float lds_e[BS][512];
  __shared__ __align__(16) bf16_t lds_attn[BS][512];
  int l0 = blockIdx.x * 16;
  int t = threadIdx.x, b = t >> 5, lane = t & 31;
  v16bf a_th = load_a16x32(thetaT + ((size_t)b * LL + l0) * 32, 32, lane);
  v8f acc[8];
  v8f zero = {};
  #pragma unroll
  for (int i = 0; i < 8; ++i) acc[i] = zero;
  const bf16_t* gb = gT + (size_t)b * (C2 * L4);
  for (int m0 = 0; m0 < L4; m0 += 32) {
    __builtin_prefetch(gb + m0 + 32, 0, 1);   // global_prefetch_b8 for next g chunk
    // energy tile [16 l][32 m] for this batch
    #pragma unroll
    for (int h = 0; h < 2; ++h) {
      v16bf Bp = load_b32x16(phiT + ((size_t)b * L4 + m0 + h * 16) * 32, 32, lane);
      v8f e = wmma_bf16(a_th, Bp, zero);
      int col = h * 16 + (lane & 15);
      int r0  = (lane < 16) ? 0 : 8;
      #pragma unroll
      for (int v = 0; v < 8; ++v) lds_e[b][(r0 + v) * 32 + col] = e[v];
    }
    __syncthreads();
    // softmax across the 8 batches, elementwise in (l,m); write attn + bf16 copy
    #pragma unroll
    for (int rep = 0; rep < 2; ++rep) {
      int idx = t + rep * 256;
      float ev[BS];
      float mx = -3.402823466e38f;
      #pragma unroll
      for (int bb = 0; bb < BS; ++bb) { ev[bb] = lds_e[bb][idx]; mx = fmaxf(mx, ev[bb]); }
      float s = 0.f;
      #pragma unroll
      for (int bb = 0; bb < BS; ++bb) { ev[bb] = __expf(ev[bb] - mx); s += ev[bb]; }
      float inv = 1.0f / s;
      int lr = idx >> 5, mr = idx & 31;
      size_t o = (size_t)(l0 + lr) * L4 + (m0 + mr);
      #pragma unroll
      for (int bb = 0; bb < BS; ++bb) {
        float a = ev[bb] * inv;
        attn_out[(size_t)bb * LL * L4 + o] = a;
        lds_attn[bb][idx] = (bf16_t)a;
      }
    }
    __syncthreads();
    // attn tile (A, K=32 m) x g (B) accumulated over 8 c'-tiles
    v16bf Aat = load_a16x32(&lds_attn[b][0], 32, lane);
    #pragma unroll
    for (int ct = 0; ct < 8; ++ct) {
      v16bf Bg = load_b32x16(gb + (size_t)ct * 16 * L4 + m0, L4, lane);
      acc[ct] = wmma_bf16(Aat, Bg, acc[ct]);
    }
    __syncthreads();
  }
  // store attn_g so the raw [C2][4096] view reads contiguously: avT[pix][c2]
  bf16_t* av = avT + (size_t)b * (LL * C2);
  #pragma unroll
  for (int ct = 0; ct < 8; ++ct) {
    int c = ct * 16 + (lane & 15);
    #pragma unroll
    for (int v = 0; v < 8; ++v) {
      int l = l0 + ((lane < 16) ? v : v + 8);
      int p = l * C2 + c;                       // flat index of [l][c] layout
      av[(size_t)(p & 4095) * C2 + (p >> 12)] = (bf16_t)acc[ct][v];
    }
  }
}

// ---- k3: last 1x1 conv (WMMA) + bias + gamma*y + x residual ----------------
__global__ void __launch_bounds__(256) k_last(const bf16_t* __restrict__ wlastb,
                                              const float* __restrict__ blast,
                                              const float* __restrict__ gamma,
                                              const bf16_t* __restrict__ avT,
                                              const float* __restrict__ x,
                                              float* __restrict__ out) {
  int b = blockIdx.x >> 6;            // 512 blocks: 8 b * 64 pixel-tiles
  int pix0 = (blockIdx.x & 63) * 64;
  int t = threadIdx.x, w = t >> 5, lane = t & 31;
  float gm = gamma[0];
  const bf16_t* av = avT + (size_t)b * (LL * C2);
  for (int mt = w * 2; mt < w * 2 + 2; ++mt) {
    int o0 = mt * 16;
    for (int nt = 0; nt < 4; ++nt) {
      int p0 = pix0 + nt * 16;
      v8f acc = {};
      #pragma unroll
      for (int kc = 0; kc < 4; ++kc) {  // K = 128
        v16bf A = load_a16x32(wlastb + (size_t)o0 * C2 + kc * 32, C2, lane);
        v16bf B = load_b32x16(av + (size_t)p0 * C2 + kc * 32, C2, lane);
        acc = wmma_bf16(A, B, acc);
      }
      int pix = p0 + (lane & 15);
      int r0  = (lane < 16) ? 0 : 8;
      #pragma unroll
      for (int v = 0; v < 8; ++v) {
        int o = o0 + r0 + v;
        size_t oi = ((size_t)b * CHN + o) * LL + pix;
        out[oi] = gm * (acc[v] + blast[o]) + x[oi];
      }
    }
  }
}

// ---- host ------------------------------------------------------------------
extern "C" void kernel_launch(void* const* d_in, const int* in_sizes, int n_in,
                              void* d_out, int out_size, void* d_ws, size_t ws_size,
                              hipStream_t stream) {
  const float* x     = (const float*)d_in[0];
  const float* wt    = (const float*)d_in[1];
  const float* bt    = (const float*)d_in[2];
  const float* wp    = (const float*)d_in[3];
  const float* bp    = (const float*)d_in[4];
  const float* wg    = (const float*)d_in[5];
  const float* bg    = (const float*)d_in[6];
  const float* wl    = (const float*)d_in[7];
  const float* bl    = (const float*)d_in[8];
  const float* gamma = (const float*)d_in[9];

  float* out      = (float*)d_out;
  float* attn_out = out + (size_t)BS * CHN * LL;   // tuple: (out, attn) concatenated

  char* w = (char*)d_ws;
  size_t o_xT     = 0;                         // 8*4096*256*2 = 16 MB
  size_t o_thetaT = o_xT + 16777216;           // 8*4096*32*2  = 2 MB
  size_t o_phiT   = o_thetaT + 2097152;        // 8*1024*32*2  = 512 KB
  size_t o_gT     = o_phiT + 524288;           // 8*128*1024*2 = 2 MB
  size_t o_avT    = o_gT + 2097152;            // 8*4096*128*2 = 8 MB
  size_t o_wcat   = o_avT + 8388608;           // 192*256*2
  size_t o_wlast  = o_wcat + 98304;            // 256*128*2
  size_t o_bcat   = o_wlast + 65536;           // 192*4
  bf16_t* xT     = (bf16_t*)(w + o_xT);
  bf16_t* thetaT = (bf16_t*)(w + o_thetaT);
  bf16_t* phiT   = (bf16_t*)(w + o_phiT);
  bf16_t* gT     = (bf16_t*)(w + o_gT);
  bf16_t* avT    = (bf16_t*)(w + o_avT);
  bf16_t* wcat   = (bf16_t*)(w + o_wcat);
  bf16_t* wlastb = (bf16_t*)(w + o_wlast);
  float*  bcat   = (float*)(w + o_bcat);

  k_transpose_x <<<8192, 256, 0, stream>>>(x, xT);
  k_prep_weights<<<1,    256, 0, stream>>>(wt, wp, wg, bt, bp, bg, wl, wcat, wlastb, bcat);
  k_proj        <<<256,  256, 0, stream>>>(xT, wcat, bcat, thetaT, phiT, gT);
  k_attn        <<<256,  256, 0, stream>>>(thetaT, phiT, gT, attn_out, avT);
  k_last        <<<512,  256, 0, stream>>>(wlastb, bl, gamma, avT, x, out);
}
